// NeRF_54752243090148
// MI455X (gfx1250) — matrix operand order repaired and checked by reading, not compiled, not run on hardware
//
#include <hip/hip_runtime.h>
#include <hip/hip_bf16.h>

typedef __attribute__((ext_vector_type(16))) _Float16 v16h;
typedef __attribute__((ext_vector_type(8)))  _Float16 v8h;
typedef __attribute__((ext_vector_type(8)))  float    v8f;

#define ROWS    64
#define MT      4        // M-tiles of 16 rows per block
#define S_ACT   304      // activation LDS stride in halves (288 used; 304*2B=608B -> conflict-free b128)
#define S_XE    72       // xe LDS stride in halves (64 used; 144B -> conflict-free b128)
#define NWAVES  8

// dynamic-LDS layout (offsets in halves)
#define L_ACTA  0
#define L_ACTB  (ROWS * S_ACT)               // 19456
#define L_XE    (2 * ROWS * S_ACT)           // 38912
#define L_DE    (L_XE + ROWS * S_XE)         // 43520
#define L_SIG   (L_DE + ROWS * 32)           // 45568 (float sigma after this, 4B aligned)
#define SMEM_BYTES (L_SIG * 2 + ROWS * 4)    // 91392 bytes

// packed f16 weight offsets (in halves) inside d_ws
#define OFF_W0   0u        // [256][64]
#define OFF_W1   16384u    // [256][256]
#define OFF_W2   81920u
#define OFF_W3   147456u
#define OFF_W4   212992u
#define OFF_W5   278528u   // [256][320]  (cols 0..63 = xe part w/ zero col 63; 64..319 = y part)
#define OFF_W6   360448u
#define OFF_W7   425984u
#define OFF_GEO  491520u   // [256][256]
#define OFF_C0   557056u   // [128][288]

struct NerfParams {
  const float* x; const float* d;
  const _Float16* W;
  const float* sb[8];
  const float* density_w; const float* density_b;
  const float* geo_b;
  const float* color_b0;
  const float* color_w1; const float* color_b1;
  float* out;
};

// f32 -> f16 weight pack with K padding; skipCol >= 0 inserts a zero column there
__global__ void pack_weights(const float* __restrict__ src, _Float16* __restrict__ dst,
                             int outDim, int inDim, int Kp, int skipCol) {
  int idx = blockIdx.x * blockDim.x + threadIdx.x;
  int total = outDim * Kp;
  if (idx >= total) return;
  int o = idx / Kp, k = idx - o * Kp;
  int s = k;
  if (skipCol >= 0) {
    if (k == skipCol) s = -1;
    else if (k > skipCol) s = k - 1;
  }
  float v = 0.f;
  if (s >= 0 && s < inDim) v = src[o * inDim + s];
  dst[idx] = (_Float16)v;
}

// A fragment (16x32 f16): lane holds row M=lane&15; 16 consecutive K halves [kh..kh+15]
__device__ __forceinline__ v16h load_a_frag(const _Float16* p) {
  v8h lo = *(const v8h*)(p);
  v8h hi = *(const v8h*)(p + 8);
  v16h a;
#pragma unroll
  for (int i = 0; i < 8; ++i) { a[i] = lo[i]; a[i + 8] = hi[i]; }
  return a;
}

// B fragment (32x16 f16): lane = output column; 16 consecutive K halves
__device__ __forceinline__ v16h load_b_frag(const _Float16* p) {
  v8h lo = *(const v8h*)(p);
  v8h hi = *(const v8h*)(p + 8);
  v16h b;
#pragma unroll
  for (int i = 0; i < 8; ++i) { b[i] = lo[i]; b[i + 8] = hi[i]; }
  return b;
}

// out[64][nTiles*16] = act( [inA | inB][64][kA+kB] * W^T + b ), two K segments share the accumulators
template<bool RELU>
__device__ __forceinline__ void gemm_layer(
    const _Float16* __restrict__ W, int Kp, const float* __restrict__ bias,
    const _Float16* inA, int strideA, int kA,
    const _Float16* inB, int strideB, int kB,
    _Float16* out, int nTiles, int lane, int wave)
{
  const int khA   = (lane < 16) ? 0 : 16;
  const int khB   = (lane < 16) ? 0 : 16;
  const int nloc  = lane & 15;
  const int rbase = (lane < 16) ? 0 : 8;
  for (int nt = wave; nt < nTiles; nt += NWAVES) {
    const int ncol = nt * 16 + nloc;
    const float bv = bias[ncol];
    const _Float16* wrow = W + (size_t)ncol * (size_t)Kp + khB;
    v8f acc[MT];
#pragma unroll
    for (int m = 0; m < MT; ++m) acc[m] = (v8f){};
    for (int kb = 0; kb < kA; kb += 32) {
      v16h bf = load_b_frag(wrow + kb);
      const _Float16* ap = inA + nloc * strideA + kb + khA;
#pragma unroll
      for (int m = 0; m < MT; ++m) {
        v16h a = load_a_frag(ap + m * 16 * strideA);
        acc[m] = __builtin_amdgcn_wmma_f32_16x16x32_f16(false, a, false, bf,
                                                        (short)0, acc[m], false, false);
      }
    }
    const _Float16* wrow2 = wrow + kA;
    for (int kb = 0; kb < kB; kb += 32) {
      v16h bf = load_b_frag(wrow2 + kb);
      const _Float16* ap = inB + nloc * strideB + kb + khA;
#pragma unroll
      for (int m = 0; m < MT; ++m) {
        v16h a = load_a_frag(ap + m * 16 * strideB);
        acc[m] = __builtin_amdgcn_wmma_f32_16x16x32_f16(false, a, false, bf,
                                                        (short)0, acc[m], false, false);
      }
    }
#pragma unroll
    for (int m = 0; m < MT; ++m) {
#pragma unroll
      for (int v = 0; v < 8; ++v) {
        float f = acc[m][v] + bv;
        if (RELU) f = fmaxf(f, 0.f);
        out[(m * 16 + rbase + v) * S_ACT + ncol] = (_Float16)f;
      }
    }
  }
}

__global__ __launch_bounds__(256) void nerf_fused(NerfParams p) {
  extern __shared__ __align__(16) _Float16 smem[];
  _Float16* actA = smem + L_ACTA;
  _Float16* actB = smem + L_ACTB;
  _Float16* xe   = smem + L_XE;
  _Float16* de   = smem + L_DE;
  float*    sig  = (float*)(smem + L_SIG);

  const int tid  = threadIdx.x;
  const int lane = tid & 31;
  const int wave = tid >> 5;
  const int row0 = blockIdx.x * ROWS;

  // ---- positional embeddings: xe (63 + zero pad), de (27).  192 threads: (row, coord)
  if (tid < 192) {
    const int r = tid / 3, ci = tid - 3 * r;
    const int gr = row0 + r;
    float xv = p.x[gr * 3 + ci];
    xe[r * S_XE + ci] = (_Float16)xv;
    float sc = 1.f;
    for (int i = 0; i < 10; ++i) {
      float arg = xv * sc; sc *= 2.f;
      xe[r * S_XE + 3 + 6 * i + ci] = (_Float16)__sinf(arg);
      xe[r * S_XE + 6 + 6 * i + ci] = (_Float16)__cosf(arg);
    }
    float dv = p.d[gr * 3 + ci];
    de[r * 32 + ci] = (_Float16)dv;
    sc = 1.f;
    for (int i = 0; i < 4; ++i) {
      float arg = dv * sc; sc *= 2.f;
      de[r * 32 + 3 + 6 * i + ci] = (_Float16)__sinf(arg);
      de[r * 32 + 6 + 6 * i + ci] = (_Float16)__cosf(arg);
    }
  }
  if (tid < ROWS) xe[tid * S_XE + 63] = (_Float16)0.f;
  __syncthreads();

  const _Float16* W = p.W;
  gemm_layer<true >(W + OFF_W0,  64,  p.sb[0], xe,   S_XE,  64,  nullptr, 0, 0,   actA, 16, lane, wave); __syncthreads();
  gemm_layer<true >(W + OFF_W1,  256, p.sb[1], actA, S_ACT, 256, nullptr, 0, 0,   actB, 16, lane, wave); __syncthreads();
  gemm_layer<true >(W + OFF_W2,  256, p.sb[2], actB, S_ACT, 256, nullptr, 0, 0,   actA, 16, lane, wave); __syncthreads();
  gemm_layer<true >(W + OFF_W3,  256, p.sb[3], actA, S_ACT, 256, nullptr, 0, 0,   actB, 16, lane, wave); __syncthreads();
  gemm_layer<true >(W + OFF_W4,  256, p.sb[4], actB, S_ACT, 256, nullptr, 0, 0,   actA, 16, lane, wave); __syncthreads();
  // skip layer: two K segments [xe(64) | y4(256)] accumulate into the same tiles
  gemm_layer<true >(W + OFF_W5,  320, p.sb[5], xe,   S_XE,  64,  actA, S_ACT, 256, actB, 16, lane, wave); __syncthreads();
  gemm_layer<true >(W + OFF_W6,  256, p.sb[6], actB, S_ACT, 256, nullptr, 0, 0,   actA, 16, lane, wave); __syncthreads();
  gemm_layer<true >(W + OFF_W7,  256, p.sb[7], actA, S_ACT, 256, nullptr, 0, 0,   actB, 16, lane, wave); __syncthreads();
  // geo (no relu): y7 (actB) -> geo_out (actA cols 0..255)
  gemm_layer<false>(W + OFF_GEO, 256, p.geo_b, actB, S_ACT, 256, nullptr, 0, 0,   actA, 16, lane, wave); __syncthreads();

  // sigma = y7 . density_w + b (reads actB, before color0 clobbers it); append de + zero pad to actA
  if (tid < ROWS) {
    float s = p.density_b[0];
    const _Float16* yr = actB + tid * S_ACT;
    for (int k = 0; k < 256; ++k) s += (float)yr[k] * p.density_w[k];
    sig[tid] = s;
  }
  for (int idx = tid; idx < ROWS * 32; idx += 256) {
    int r = idx >> 5, c = idx & 31;
    actA[r * S_ACT + 256 + c] = (c < 27) ? de[idx] : (_Float16)0.f;
  }
  __syncthreads();

  // color0: [64][288] -> relu -> actB cols 0..127
  gemm_layer<true>(W + OFF_C0, 288, p.color_b0, actA, S_ACT, 288, nullptr, 0, 0, actB, 8, lane, wave);
  __syncthreads();

  // final: rgb = c0 . color_w1^T + b1, out[:,3] = sigma   (64 rows x 4 outputs = 256 threads)
  {
    int r = tid >> 2, c = tid & 3;
    int gr = row0 + r;
    if (c < 3) {
      float s = p.color_b1[c];
      const _Float16* cr = actB + r * S_ACT;
      for (int k = 0; k < 128; ++k) s += (float)cr[k] * p.color_w1[c * 128 + k];
      p.out[gr * 4 + c] = s;
    } else {
      p.out[gr * 4 + 3] = sig[r];
    }
  }
}

extern "C" void kernel_launch(void* const* d_in, const int* in_sizes, int n_in,
                              void* d_out, int out_size, void* d_ws, size_t ws_size,
                              hipStream_t stream) {
  _Float16* W = (_Float16*)d_ws;

  auto packLaunch = [&](const void* src, _Float16* dst, int oD, int iD, int Kp, int skip) {
    int total = oD * Kp;
    pack_weights<<<(total + 255) / 256, 256, 0, stream>>>((const float*)src, dst, oD, iD, Kp, skip);
  };
  packLaunch(d_in[2],  W + OFF_W0,  256, 63,  64,  -1);
  packLaunch(d_in[4],  W + OFF_W1,  256, 256, 256, -1);
  packLaunch(d_in[6],  W + OFF_W2,  256, 256, 256, -1);
  packLaunch(d_in[8],  W + OFF_W3,  256, 256, 256, -1);
  packLaunch(d_in[10], W + OFF_W4,  256, 256, 256, -1);
  packLaunch(d_in[12], W + OFF_W5,  256, 319, 320, 63);  // zero col at k=63 (xe pad slot)
  packLaunch(d_in[14], W + OFF_W6,  256, 256, 256, -1);
  packLaunch(d_in[16], W + OFF_W7,  256, 256, 256, -1);
  packLaunch(d_in[20], W + OFF_GEO, 256, 256, 256, -1);
  packLaunch(d_in[22], W + OFF_C0,  128, 283, 288, -1);

  NerfParams P;
  P.x = (const float*)d_in[0];
  P.d = (const float*)d_in[1];
  P.W = W;
  for (int i = 0; i < 8; ++i) P.sb[i] = (const float*)d_in[3 + 2 * i];
  P.density_w = (const float*)d_in[18];
  P.density_b = (const float*)d_in[19];
  P.geo_b     = (const float*)d_in[21];
  P.color_b0  = (const float*)d_in[23];
  P.color_w1  = (const float*)d_in[24];
  P.color_b1  = (const float*)d_in[25];
  P.out = (float*)d_out;

  int N = in_sizes[0] / 3;
  nerf_fused<<<N / ROWS, 256, SMEM_BYTES, stream>>>(P);
}
